// NonLocalBlock2D_37031208026178
// MI455X (gfx1250) — matrix operand order repaired and checked by reading, not compiled, not run on hardware
//
#include <hip/hip_runtime.h>
#include <cstdint>

#define B_  8
#define C_  256
#define CI_ 128
#define HW_ 4096   // N = 64*64

typedef __attribute__((ext_vector_type(16))) __bf16 v16bf;
typedef __attribute__((ext_vector_type(8)))  float  v8f;
typedef int vi4 __attribute__((vector_size(4 * sizeof(int))));
typedef vi4 __attribute__((address_space(1)))* as1_vi4p;   // global
typedef vi4 __attribute__((address_space(3)))* as3_vi4p;   // LDS

union BF16x16 { v16bf v; unsigned u[8]; };

#if __has_builtin(__builtin_amdgcn_global_load_async_to_lds_b128)
#define HAVE_ASYNC_LDS 1
#else
#define HAVE_ASYNC_LDS 0
#endif

__device__ __forceinline__ unsigned short f2bf(float f) {
  unsigned x = __float_as_uint(f);
  unsigned r = (x + 0x7FFFu + ((x >> 16) & 1u)) >> 16;   // RNE f32 -> bf16
  return (unsigned short)r;
}
__device__ __forceinline__ unsigned pack2(float a, float b) {
  return (unsigned)f2bf(a) | ((unsigned)f2bf(b) << 16);
}
__device__ __forceinline__ v8f wmma_bf16(const BF16x16& a, const BF16x16& b, v8f c) {
  return __builtin_amdgcn_wmma_f32_16x16x32_bf16(false, a.v, false, b.v,
                                                 (short)0, c, false, false);
}
// A-fragment K offset, contiguous order (k = 16h + 2j): dword j, lane-half h
__device__ __forceinline__ int aoff(int j, int h) {
  return h * 16 + 2 * j;
}

// 16-byte global -> LDS copy: async path (ASYNCcnt-tracked) if available.
__device__ __forceinline__ void cp16(unsigned short* lds, const unsigned short* g) {
#if HAVE_ASYNC_LDS
  __builtin_amdgcn_global_load_async_to_lds_b128((as1_vi4p)g, (as3_vi4p)lds, 0, 0);
#else
  *(uint4*)lds = *(const uint4*)g;
#endif
}
__device__ __forceinline__ void stage_commit() {
#if HAVE_ASYNC_LDS
#if __has_builtin(__builtin_amdgcn_s_wait_asynccnt)
  __builtin_amdgcn_s_wait_asynccnt(0);
#else
  asm volatile("s_wait_asynccnt 0" ::: "memory");
#endif
#endif
  __syncthreads();
}

// ---------------------------------------------------------------------------
// Kernel 1: three 1x1-conv projections.  out = W[CI,C] @ x[C,N] + bias
// theta/phi -> bf16 [B][N][CI] ; g -> bf16 [B][CI][N]
// ---------------------------------------------------------------------------
__global__ __launch_bounds__(256) void proj_kernel(
    const float* __restrict__ x,
    const float* __restrict__ w_theta, const float* __restrict__ b_theta,
    const float* __restrict__ w_phi,   const float* __restrict__ b_phi,
    const float* __restrict__ w_g,     const float* __restrict__ b_g,
    unsigned short* __restrict__ thetaT,
    unsigned short* __restrict__ phiT,
    unsigned short* __restrict__ gCN)
{
  const int nblock = blockIdx.x * 32;
  const int proj   = blockIdx.y;        // 0=theta, 1=phi, 2=g
  const int b      = blockIdx.z;

  const float* W; const float* bias;
  if (proj == 0)      { W = w_theta; bias = b_theta; }
  else if (proj == 1) { W = w_phi;   bias = b_phi;   }
  else                { W = w_g;     bias = b_g;     }

  __shared__ unsigned short xT[32 * 32];   // x tile transposed: [n_local][c_local] bf16

  const int tid  = threadIdx.x;
  const int wid  = tid >> 5;
  const int lane = tid & 31;
  const int l16  = lane & 15;
  const int half = lane >> 4;

  v8f acc0 = {}; v8f acc1 = {};
  const float* xb = x + (size_t)b * C_ * HW_;

  for (int cb = 0; cb < C_; cb += 32) {
    __syncthreads();
    #pragma unroll
    for (int i = 0; i < 4; ++i) {
      int idx = tid + i * 256;
      int cl = idx >> 5, nl = idx & 31;
      float v = xb[(size_t)(cb + cl) * HW_ + nblock + nl];
      xT[nl * 32 + cl] = f2bf(v);
    }
    __syncthreads();

    BF16x16 a;
    const float* wr = W + (size_t)(wid * 16 + l16) * C_ + cb;
    #pragma unroll
    for (int j = 0; j < 8; ++j) {
      int co = aoff(j, half);
      a.u[j] = pack2(wr[co], wr[co + 1]);
    }
    BF16x16 b0, b1;
    #pragma unroll
    for (int j = 0; j < 8; ++j) {
      b0.u[j] = *(const unsigned*)&xT[l16 * 32        + half * 16 + 2 * j];
      b1.u[j] = *(const unsigned*)&xT[(16 + l16) * 32 + half * 16 + 2 * j];
    }
    acc0 = wmma_bf16(a, b0, acc0);
    acc1 = wmma_bf16(a, b1, acc1);
  }

  #pragma unroll
  for (int r = 0; r < 8; ++r) {
    int o = wid * 16 + r + 8 * half;
    float bv = bias[o];
    float v0 = acc0[r] + bv, v1 = acc1[r] + bv;
    int n0 = nblock + l16, n1 = nblock + 16 + l16;
    if (proj == 2) {
      gCN[((size_t)b * CI_ + o) * HW_ + n0] = f2bf(v0);
      gCN[((size_t)b * CI_ + o) * HW_ + n1] = f2bf(v1);
    } else {
      unsigned short* dst = (proj == 0) ? thetaT : phiT;
      dst[((size_t)b * HW_ + n0) * CI_ + o] = f2bf(v0);
      dst[((size_t)b * HW_ + n1) * CI_ + o] = f2bf(v1);
    }
  }
}

// ---------------------------------------------------------------------------
// Kernel 2: flash attention.  pw = softmax(sqrt(CI) * theta @ phi^T) ; y = pw @ g
// 4 waves / block. Per 32-key block, phi & g tiles are staged ONCE into LDS
// (async global->LDS when available), then all waves feed WMMA from LDS.
// ---------------------------------------------------------------------------
__global__ __launch_bounds__(128) void attn_kernel(
    const unsigned short* __restrict__ thetaT,  // [B][N][CI]
    const unsigned short* __restrict__ phiT,    // [B][N][CI]
    const unsigned short* __restrict__ gCN,     // [B][CI][N]
    unsigned short* __restrict__ yT)            // [B][N][CI]
{
  const int b    = blockIdx.y;
  const int tid  = threadIdx.x;
  const int wid  = tid >> 5;
  const int lane = tid & 31;
  const int l16  = lane & 15;
  const int half = lane >> 4;
  const int qbase = blockIdx.x * 64 + wid * 16;

  __shared__ unsigned short Kt[32 * 128];       // phi tile  [key][ci]   8 KB
  __shared__ unsigned short Gt[128 * 32];       // g tile    [ci][key]   8 KB
  __shared__ unsigned short Plds[4][16 * 32];   // wave-private P relayout tiles
  unsigned short* pbuf = Plds[wid];

  const float scale = 11.313708498984760f;      // sqrt(CI)

  // Preload Q: four 16x32 A-fragments over CI
  BF16x16 qf[4];
  {
    const unsigned short* qrow = thetaT + ((size_t)b * HW_ + qbase + l16) * CI_;
    #pragma unroll
    for (int c = 0; c < 4; ++c)
      #pragma unroll
      for (int j = 0; j < 8; ++j)
        qf[c].u[j] = *(const unsigned*)&qrow[c * 32 + aoff(j, half)];
  }

  v8f O[8];
  v8f vz = {};
  #pragma unroll
  for (int c = 0; c < 8; ++c) O[c] = vz;
  float m[8], l[8];
  #pragma unroll
  for (int r = 0; r < 8; ++r) { m[r] = -3.0e38f; l[r] = 0.f; }

  const unsigned short* phb = phiT + (size_t)b * HW_ * CI_;
  const unsigned short* gb  = gCN  + (size_t)b * CI_ * HW_;

  for (int kb = 0; kb < HW_; kb += 32) {
    // ---- stage phi tile: contiguous 8 KB (rows kb..kb+31 of phi^T) ----
    const unsigned short* ksrc = phb + (size_t)kb * CI_;
    #pragma unroll
    for (int k = 0; k < 4; ++k) {
      int f = (tid + k * 128) * 8;              // element offset, 8 bf16 per b128
      cp16(&Kt[f], &ksrc[f]);
    }
    // ---- stage g tile: 128 rows x 64 B from g[CI][N] ----
    #pragma unroll
    for (int k = 0; k < 4; ++k) {
      int idx = tid + k * 128;                  // 512 x b128
      int row = idx >> 2, quad = idx & 3;
      cp16(&Gt[row * 32 + quad * 8], &gb[(size_t)row * HW_ + kb + quad * 8]);
    }
    stage_commit();

    // ---- S = Q @ K^T  (16q x 32k, 8 WMMA) ----
    v8f s0 = vz, s1 = vz;
    #pragma unroll
    for (int c = 0; c < 4; ++c) {
      BF16x16 k0, k1;
      const unsigned short* kr0 = &Kt[l16 * CI_        + c * 32 + half * 16];
      const unsigned short* kr1 = &Kt[(16 + l16) * CI_ + c * 32 + half * 16];
      #pragma unroll
      for (int j = 0; j < 8; ++j) {
        k0.u[j] = *(const unsigned*)&kr0[2 * j];
        k1.u[j] = *(const unsigned*)&kr1[2 * j];
      }
      s0 = wmma_bf16(qf[c], k0, s0);
      s1 = wmma_bf16(qf[c], k1, s1);
    }
    // ---- online softmax over this 32-key slab ----
    #pragma unroll
    for (int r = 0; r < 8; ++r) {
      float a0 = s0[r] * scale, a1 = s1[r] * scale;
      float mx = fmaxf(a0, a1);
      #pragma unroll
      for (int off = 1; off < 16; off <<= 1) mx = fmaxf(mx, __shfl_xor(mx, off, 32));
      float mn = fmaxf(m[r], mx);
      float p0 = __expf(a0 - mn), p1 = __expf(a1 - mn);
      float al = __expf(m[r] - mn);
      m[r] = mn;
      float rs = p0 + p1;
      #pragma unroll
      for (int off = 1; off < 16; off <<= 1) rs += __shfl_xor(rs, off, 32);
      l[r] = l[r] * al + rs;
      #pragma unroll
      for (int c = 0; c < 8; ++c) O[c][r] *= al;
      pbuf[(r + 8 * half) * 32      + l16] = f2bf(p0);
      pbuf[(r + 8 * half) * 32 + 16 + l16] = f2bf(p1);
    }
    // ---- P (A-layout, via in-order wave-private LDS) ----
    BF16x16 pA;
    #pragma unroll
    for (int j = 0; j < 8; ++j)
      pA.u[j] = *(const unsigned*)&pbuf[l16 * 32 + aoff(j, half)];
    // ---- O += P @ G  (16q x 128ci, 8 WMMA) ----
    #pragma unroll
    for (int c = 0; c < 8; ++c) {
      BF16x16 gB;
      const unsigned short* gr = &Gt[(c * 16 + l16) * 32 + half * 16];
      #pragma unroll
      for (int j = 0; j < 8; ++j) gB.u[j] = *(const unsigned*)&gr[2 * j];
      O[c] = wmma_bf16(pA, gB, O[c]);
    }
    __syncthreads();                            // tiles fully consumed before re-stage
  }

  #pragma unroll
  for (int r = 0; r < 8; ++r) {
    float inv = 1.0f / l[r];
    int q = qbase + r + 8 * half;
    #pragma unroll
    for (int c = 0; c < 8; ++c)
      yT[((size_t)b * HW_ + q) * CI_ + c * 16 + l16] = f2bf(O[c][r] * inv);
  }
}

// ---------------------------------------------------------------------------
// Kernel 3: out = x + W_out[C,CI] @ y + b_out.  8 waves as 2x4 -> 32o x 128n tile.
// ---------------------------------------------------------------------------
__global__ __launch_bounds__(256) void out_kernel(
    const float* __restrict__ x,
    const float* __restrict__ w_out, const float* __restrict__ b_out,
    const unsigned short* __restrict__ yT,      // [B][N][CI]
    float* __restrict__ out)
{
  const int b    = blockIdx.z;
  const int tid  = threadIdx.x;
  const int wid  = tid >> 5;
  const int lane = tid & 31;
  const int l16  = lane & 15;
  const int half = lane >> 4;
  const int obase = blockIdx.y * 32  + (wid >> 2) * 16;
  const int nbase = blockIdx.x * 128 + (wid & 3) * 32;

  v8f acc0 = {}, acc1 = {};
  const unsigned short* yb = yT + (size_t)b * HW_ * CI_;

  #pragma unroll
  for (int cb = 0; cb < CI_; cb += 32) {
    BF16x16 a;
    const float* wr = w_out + (size_t)(obase + l16) * CI_ + cb;
    #pragma unroll
    for (int j = 0; j < 8; ++j) {
      int co = aoff(j, half);
      a.u[j] = pack2(wr[co], wr[co + 1]);
    }
    BF16x16 b0, b1;
    const unsigned short* y0 = yb + (size_t)(nbase + l16)      * CI_ + cb + half * 16;
    const unsigned short* y1 = yb + (size_t)(nbase + 16 + l16) * CI_ + cb + half * 16;
    #pragma unroll
    for (int j = 0; j < 8; ++j) {
      b0.u[j] = *(const unsigned*)&y0[2 * j];
      b1.u[j] = *(const unsigned*)&y1[2 * j];
    }
    acc0 = wmma_bf16(a, b0, acc0);
    acc1 = wmma_bf16(a, b1, acc1);
  }

  #pragma unroll
  for (int r = 0; r < 8; ++r) {
    int o = obase + r + 8 * half;
    float bv = b_out[o];
    size_t base = ((size_t)b * C_ + o) * HW_;
    int n0 = nbase + l16, n1 = nbase + 16 + l16;
    out[base + n0] = acc0[r] + bv + x[base + n0];
    out[base + n1] = acc1[r] + bv + x[base + n1];
  }
}

// ---------------------------------------------------------------------------
extern "C" void kernel_launch(void* const* d_in, const int* in_sizes, int n_in,
                              void* d_out, int out_size, void* d_ws, size_t ws_size,
                              hipStream_t stream) {
  const float* x       = (const float*)d_in[0];
  const float* w_g     = (const float*)d_in[1];
  const float* b_g     = (const float*)d_in[2];
  const float* w_theta = (const float*)d_in[3];
  const float* b_theta = (const float*)d_in[4];
  const float* w_phi   = (const float*)d_in[5];
  const float* b_phi   = (const float*)d_in[6];
  const float* w_out   = (const float*)d_in[7];
  const float* b_out   = (const float*)d_in[8];

  unsigned short* ws = (unsigned short*)d_ws;
  const size_t SZ = (size_t)B_ * HW_ * CI_;     // 4 Mi elems each, 32 MB total
  unsigned short* thetaT = ws;
  unsigned short* phiT   = ws +     SZ;
  unsigned short* gCN    = ws + 2 * SZ;
  unsigned short* yT     = ws + 3 * SZ;

  proj_kernel<<<dim3(HW_ / 32, 3, B_), 256, 0, stream>>>(
      x, w_theta, b_theta, w_phi, b_phi, w_g, b_g, thetaT, phiT, gCN);
  attn_kernel<<<dim3(HW_ / 64, B_), 128, 0, stream>>>(thetaT, phiT, gCN, yT);
  out_kernel<<<dim3(HW_ / 128, C_ / 32, B_), 256, 0, stream>>>(
      x, w_out, b_out, yT, (float*)d_out);
}
